// Attention_33861522162538
// MI455X (gfx1250) — matrix operand order repaired and checked by reading, not compiled, not run on hardware
//
#include <hip/hip_runtime.h>
#include <hip/hip_bf16.h>

typedef __attribute__((ext_vector_type(16))) _Float16 v16h;
typedef __attribute__((ext_vector_type(8)))  _Float16 v8h;
typedef __attribute__((ext_vector_type(8)))  float    v8f;
typedef __attribute__((ext_vector_type(8)))  unsigned u8v;

#define BB 2
#define LL 2048
#define DD 1024
#define HH 16
#define DH 64
#define BH 32
#define MM 4096           // B*L
#define SCALE_F 0.125f    // DH^-0.5

__device__ __forceinline__ v8f wmma16(v16h a, v16h b, v8f c) {
  // D = A(16x32 f16) * B(32x16 f16) + C(16x16 f32)
  return __builtin_amdgcn_wmma_f32_16x16x32_f16(false, a, false, b, (short)0, c, false, false);
}

__device__ __forceinline__ unsigned pack2h(float a, float b) {
  unsigned short ua = __builtin_bit_cast(unsigned short, (_Float16)a);
  unsigned short ub = __builtin_bit_cast(unsigned short, (_Float16)b);
  return (unsigned)ua | ((unsigned)ub << 16);
}

__device__ __forceinline__ v16h cvt16(const float4* src) {
  float4 f0 = src[0], f1 = src[1], f2 = src[2], f3 = src[3];
  v16h hv;
  hv[0]  = (_Float16)f0.x; hv[1]  = (_Float16)f0.y; hv[2]  = (_Float16)f0.z; hv[3]  = (_Float16)f0.w;
  hv[4]  = (_Float16)f1.x; hv[5]  = (_Float16)f1.y; hv[6]  = (_Float16)f1.z; hv[7]  = (_Float16)f1.w;
  hv[8]  = (_Float16)f2.x; hv[9]  = (_Float16)f2.y; hv[10] = (_Float16)f2.z; hv[11] = (_Float16)f2.w;
  hv[12] = (_Float16)f3.x; hv[13] = (_Float16)f3.y; hv[14] = (_Float16)f3.z; hv[15] = (_Float16)f3.w;
  return hv;
}

__device__ __forceinline__ v8h cvt8(float4 f0, float4 f1) {
  v8h hv;
  hv[0] = (_Float16)f0.x; hv[1] = (_Float16)f0.y; hv[2] = (_Float16)f0.z; hv[3] = (_Float16)f0.w;
  hv[4] = (_Float16)f1.x; hv[5] = (_Float16)f1.y; hv[6] = (_Float16)f1.z; hv[7] = (_Float16)f1.w;
  return hv;
}

// ---------------------------------------------------------------------------
// Kernel 1: QKV projection.  out[e] = sum_d x[row,d] * W[e,d]  (f32 -> f16)
// Block tile 128(M) x 64(N), K-step 32, double-buffered LDS (1 barrier/step),
// 8 waves, wave does 2x2 WMMA tiles.  Output split-head: [B*H, L, DH] f16.
// ---------------------------------------------------------------------------
__global__ __launch_bounds__(256) void qkv_gemm(
    const float* __restrict__ x,
    const float* __restrict__ Wq, const float* __restrict__ Wk,
    const float* __restrict__ Wv,
    _Float16* __restrict__ Qh, _Float16* __restrict__ Kh,
    _Float16* __restrict__ Vh)
{
  __shared__ _Float16 Al[2][128 * 32];
  __shared__ _Float16 Bl[2][64 * 32];

  const int t    = threadIdx.x;
  const int lane = t & 31;
  const int w    = t >> 5;
  const int g    = lane >> 4;     // half-wave
  const int ln   = lane & 15;
  const int wm   = w >> 1;        // 0..3
  const int wn   = w & 1;         // 0..1

  const int mt = blockIdx.x >> 4;       // 32 M-tiles
  const int nt = blockIdx.x & 15;       // 16 N-tiles
  const int m0 = mt * 128;
  const int n0 = nt * 64;

  const float* W   = (blockIdx.z == 0) ? Wq : (blockIdx.z == 1 ? Wk : Wv);
  _Float16*    Out = (blockIdx.z == 0) ? Qh : (blockIdx.z == 1 ? Kh : Vh);

  const int arow  = t >> 1;
  const int acolb = (t & 1) * 16;
  const int brow  = t >> 2;
  const int bcolb = (t & 3) * 8;

  const float* srcA = x + (size_t)(m0 + arow) * DD + acolb;
  const float* srcB = W + (size_t)(n0 + brow) * DD + bcolb;

  v8f acc[2][2] = {};

  // prologue: stage K-step 0 into buffer 0
  {
    v16h ah = cvt16((const float4*)srcA);
    v8h  bh2 = cvt8(((const float4*)srcB)[0], ((const float4*)srcB)[1]);
    *(v16h*)(Al[0] + arow * 32 + acolb) = ah;
    *(v8h*) (Bl[0] + brow * 32 + bcolb) = bh2;
  }
  __syncthreads();

  const int NIT = DD / 32;
  for (int it = 0; it < NIT; ++it) {
    const int  buf  = it & 1;
    const bool more = (it + 1) < NIT;

    // load next tile to registers (overlaps with compute below)
    float4 ar0, ar1, ar2, ar3, br0, br1;
    if (more) {
      const float4* pa = (const float4*)(srcA + (it + 1) * 32);
      ar0 = pa[0]; ar1 = pa[1]; ar2 = pa[2]; ar3 = pa[3];
      const float4* pb = (const float4*)(srcB + (it + 1) * 32);
      br0 = pb[0]; br1 = pb[1];
    }

    // compute current tile from LDS
    v16h af[2], bf[2];
    #pragma unroll
    for (int mi = 0; mi < 2; ++mi) {
      const _Float16* p = Al[buf] + (wm * 32 + mi * 16 + ln) * 32 + g * 16;
      v8h lo = *(const v8h*)(p);
      v8h hi = *(const v8h*)(p + 8);
      af[mi] = __builtin_shufflevector(lo, hi, 0,1,2,3,4,5,6,7,8,9,10,11,12,13,14,15);
    }
    #pragma unroll
    for (int ni = 0; ni < 2; ++ni)
      bf[ni] = *(const v16h*)(Bl[buf] + (wn * 32 + ni * 16 + ln) * 32 + g * 16);
    #pragma unroll
    for (int mi = 0; mi < 2; ++mi)
      #pragma unroll
      for (int ni = 0; ni < 2; ++ni)
        acc[mi][ni] = wmma16(af[mi], bf[ni], acc[mi][ni]);

    // store next tile to the other LDS buffer
    if (more) {
      v16h ah;
      ah[0]  = (_Float16)ar0.x; ah[1]  = (_Float16)ar0.y; ah[2]  = (_Float16)ar0.z; ah[3]  = (_Float16)ar0.w;
      ah[4]  = (_Float16)ar1.x; ah[5]  = (_Float16)ar1.y; ah[6]  = (_Float16)ar1.z; ah[7]  = (_Float16)ar1.w;
      ah[8]  = (_Float16)ar2.x; ah[9]  = (_Float16)ar2.y; ah[10] = (_Float16)ar2.z; ah[11] = (_Float16)ar2.w;
      ah[12] = (_Float16)ar3.x; ah[13] = (_Float16)ar3.y; ah[14] = (_Float16)ar3.z; ah[15] = (_Float16)ar3.w;
      *(v16h*)(Al[buf ^ 1] + arow * 32 + acolb) = ah;
      *(v8h*) (Bl[buf ^ 1] + brow * 32 + bcolb) = cvt8(br0, br1);
    }
    __syncthreads();
  }

  // epilogue: split-head f16 store  [(b*H + h), l, dh]
  #pragma unroll
  for (int mi = 0; mi < 2; ++mi) {
    #pragma unroll
    for (int ni = 0; ni < 2; ++ni) {
      const int e  = n0 + wn * 32 + ni * 16 + ln;
      const int h  = e >> 6;
      const int dh = e & 63;
      #pragma unroll
      for (int v = 0; v < 8; ++v) {
        const int row = m0 + wm * 32 + mi * 16 + v + 8 * g; // global M row
        const int b   = row >> 11;
        const int l   = row & 2047;
        Out[((size_t)(b * HH + h) * LL + l) * DH + dh] = (_Float16)acc[mi][ni][v];
      }
    }
  }
}

// ---------------------------------------------------------------------------
// Kernel 2: fused flash attention (transposed-score formulation).
// Block: 8 waves x 16 q-rows = 128 q rows of one (batch*head).
// Double-buffered LDS tiles; K row-major, V transposed; S^T = K*Q^T via WMMA,
// online softmax (per-q stats per-lane), P rebuilt in registers via one
// half-wave shuffle, O^T = V^T * P^T via WMMA.  SCALE pre-folded into Q.
// ---------------------------------------------------------------------------
__global__ __launch_bounds__(256) void attn_kernel(
    const _Float16* __restrict__ Qh, const _Float16* __restrict__ Kh,
    const _Float16* __restrict__ Vh, const float* __restrict__ mask,
    _Float16* __restrict__ Hh)
{
  __shared__ _Float16 Kl[2][32 * 64];   // K tile, row-major [k][dh]
  __shared__ _Float16 Vt[2][64 * 32];   // V tile, transposed [dh][k]

  const int t    = threadIdx.x;
  const int lane = t & 31;
  const int w    = t >> 5;
  const int g    = lane >> 4;
  const int ln   = lane & 15;

  const int bh    = blockIdx.y;
  const int qrow0 = blockIdx.x * 128 + w * 16;
  const int ql    = qrow0 + ln;   // this lane's query row (column of S^T)

  const _Float16* Kbase = Kh + (size_t)bh * LL * DH;
  const _Float16* Vbase = Vh + (size_t)bh * LL * DH;
  const float*    mrow  = mask + ((size_t)bh * LL + ql) * LL;

  // Q as B-operand fragments (col = q, K-dim = dh), pre-scaled by 2^-3 (exact)
  const _Float16* qbase = Qh + ((size_t)bh * LL + ql) * DH;
  v16h bq0 = *(const v16h*)(qbase + g * 16);        // dh 0..31
  v16h bq1 = *(const v16h*)(qbase + 32 + g * 16);   // dh 32..63
  #pragma unroll
  for (int i = 0; i < 16; ++i) {
    bq0[i] = bq0[i] * (_Float16)SCALE_F;
    bq1[i] = bq1[i] * (_Float16)SCALE_F;
  }

  const int krow = t >> 3;        // V stage: source row within 32-key tile
  const int c0   = (t & 7) * 8;   // V stage: dh base

  float m_row = -1e30f, l_row = 0.0f;
  v8f acc[4] = {};   // O^T tiles: dh tile dt, rows dh = dt*16 + v + 8g, col q = ln

  // prologue: stage key-block 0 into buffer 0
  {
    uint4 kreg = ((const uint4*)Kbase)[t];
    uint4 vreg = ((const uint4*)Vbase)[t];
    ((uint4*)Kl[0])[t] = kreg;
    v8h hv = __builtin_bit_cast(v8h, vreg);
    #pragma unroll
    for (int i = 0; i < 8; ++i) Vt[0][(c0 + i) * 32 + krow] = hv[i];
  }
  __syncthreads();

  const int NIT = LL / 32;
  for (int it = 0; it < NIT; ++it) {
    const int  buf  = it & 1;
    const bool more = (it + 1) < NIT;

    // load next K/V tile into registers; prefetch next mask lines
    uint4 kreg, vreg;
    if (more) {
      kreg = ((const uint4*)(Kbase + (size_t)(it + 1) * 32 * DH))[t];
      vreg = ((const uint4*)(Vbase + (size_t)(it + 1) * 32 * DH))[t];
      __builtin_prefetch(mrow + (it + 1) * 32 + g * 16, 0, 0);
    }

    // ---- scores S^T for two 16-key subtiles ----
    float s[16];
    #pragma unroll
    for (int sub = 0; sub < 2; ++sub) {
      const _Float16* kp = Kl[buf] + (sub * 16 + ln) * 64 + g * 16;
      v8h c0v = *(const v8h*)(kp +  0);
      v8h c1v = *(const v8h*)(kp +  8);
      v8h c2v = *(const v8h*)(kp + 32);
      v8h c3v = *(const v8h*)(kp + 40);
      v16h a0 = __builtin_shufflevector(c0v, c1v, 0,1,2,3,4,5,6,7,8,9,10,11,12,13,14,15);
      v16h a1 = __builtin_shufflevector(c2v, c3v, 0,1,2,3,4,5,6,7,8,9,10,11,12,13,14,15);
      v8f sc = {};
      sc = wmma16(a0, bq0, sc);
      sc = wmma16(a1, bq1, sc);
      // keys covered by this lane: it*32 + sub*16 + 8g + v  (v contiguous)
      const float4* mp = (const float4*)(mrow + it * 32 + sub * 16 + 8 * g);
      float4 mk0 = mp[0], mk1 = mp[1];
      s[sub*8+0] = sc[0] + mk0.x;
      s[sub*8+1] = sc[1] + mk0.y;
      s[sub*8+2] = sc[2] + mk0.z;
      s[sub*8+3] = sc[3] + mk0.w;
      s[sub*8+4] = sc[4] + mk1.x;
      s[sub*8+5] = sc[5] + mk1.y;
      s[sub*8+6] = sc[6] + mk1.z;
      s[sub*8+7] = sc[7] + mk1.w;
    }

    // ---- online softmax update (stats per q == per lane, replicated g<->g^1)
    float bm = s[0];
    #pragma unroll
    for (int i = 1; i < 16; ++i) bm = fmaxf(bm, s[i]);
    bm = fmaxf(bm, __shfl_xor(bm, 16, 32));
    const float m_new = fmaxf(m_row, bm);
    float p[16], tsum = 0.0f;
    #pragma unroll
    for (int i = 0; i < 16; ++i) { p[i] = __expf(s[i] - m_new); tsum += p[i]; }
    tsum += __shfl_xor(tsum, 16, 32);
    const float corr = __expf(m_row - m_new);
    l_row = l_row * corr + tsum;
    m_row = m_new;
    #pragma unroll
    for (int dt = 0; dt < 4; ++dt)
      #pragma unroll
      for (int v = 0; v < 8; ++v) acc[dt][v] *= corr;

    // ---- rebuild P^T as B-operand (32 keys x 16 q) in registers ----
    unsigned pk0[4], pk1[4];
    #pragma unroll
    for (int i = 0; i < 4; ++i) {
      pk0[i] = pack2h(p[2*i],     p[2*i + 1]);     // subtile 0, my 8 keys
      pk1[i] = pack2h(p[8 + 2*i], p[9 + 2*i]);     // subtile 1, my 8 keys
    }
    unsigned keep[4], recv[4];
    #pragma unroll
    for (int i = 0; i < 4; ++i) {
      unsigned send = g ? pk0[i] : pk1[i];   // partner needs my other subtile
      keep[i] = g ? pk1[i] : pk0[i];
      recv[i] = __shfl_xor(send, 16, 32);
    }
    u8v bpu;
    #pragma unroll
    for (int i = 0; i < 4; ++i) {
      bpu[i]     = g ? recv[i] : keep[i];  // elements j=0..7  : keys g*16+0..7
      bpu[4 + i] = g ? keep[i] : recv[i];  // elements j=8..15 : keys g*16+8..15
    }
    const v16h bP = __builtin_bit_cast(v16h, bpu);

    // ---- O^T += V^T * P^T  (4 dh tiles), V^T frags are contiguous LDS b128s
    #pragma unroll
    for (int dt = 0; dt < 4; ++dt) {
      const _Float16* vp = Vt[buf] + (dt * 16 + ln) * 32 + g * 16;
      v8h lo = *(const v8h*)(vp);
      v8h hi = *(const v8h*)(vp + 8);
      v16h av = __builtin_shufflevector(lo, hi, 0,1,2,3,4,5,6,7,8,9,10,11,12,13,14,15);
      acc[dt] = wmma16(av, bP, acc[dt]);
    }

    // store next tile into the other buffer
    if (more) {
      ((uint4*)Kl[buf ^ 1])[t] = kreg;
      v8h hv = __builtin_bit_cast(v8h, vreg);
      #pragma unroll
      for (int i = 0; i < 8; ++i) Vt[buf ^ 1][(c0 + i) * 32 + krow] = hv[i];
    }
    __syncthreads();
  }

  // finalize: hidden[b, l, h*64 + dh] f16
  const float inv_l = 1.0f / l_row;
  const int b = bh >> 4, h = bh & 15;
  _Float16* out = Hh + ((size_t)(b * LL + ql)) * (HH * DH) + h * DH;
  #pragma unroll
  for (int dt = 0; dt < 4; ++dt) {
    v8h ov;
    #pragma unroll
    for (int v = 0; v < 8; ++v) ov[v] = (_Float16)(acc[dt][v] * inv_l);
    *(v8h*)(out + dt * 16 + g * 8) = ov;   // dh = dt*16 + 8g + v, contiguous
  }
}

// ---------------------------------------------------------------------------
// Kernel 3: out projection.  out[row,d] = sum_e H[row,e]*Wo[d,e] + bo[d] (f32)
// Double-buffered like kernel 1; A already f16.
// ---------------------------------------------------------------------------
__global__ __launch_bounds__(256) void out_gemm(
    const _Float16* __restrict__ Hh, const float* __restrict__ Wo,
    const float* __restrict__ bo, float* __restrict__ out)
{
  __shared__ _Float16 Al[2][128 * 32];
  __shared__ _Float16 Bl[2][64 * 32];

  const int t    = threadIdx.x;
  const int lane = t & 31;
  const int w    = t >> 5;
  const int g    = lane >> 4;
  const int ln   = lane & 15;
  const int wm   = w >> 1;
  const int wn   = w & 1;

  const int mt = blockIdx.x >> 4;
  const int nt = blockIdx.x & 15;
  const int m0 = mt * 128;
  const int n0 = nt * 64;

  const int arow  = t >> 1;
  const int acolb = (t & 1) * 16;
  const int brow  = t >> 2;
  const int bcolb = (t & 3) * 8;

  const _Float16* srcA = Hh + (size_t)(m0 + arow) * DD + acolb;
  const float*    srcB = Wo + (size_t)(n0 + brow) * DD + bcolb;

  v8f acc[2][2] = {};

  { // prologue stage
    *(v16h*)(Al[0] + arow * 32 + acolb) = *(const v16h*)srcA;
    *(v8h*) (Bl[0] + brow * 32 + bcolb) = cvt8(((const float4*)srcB)[0], ((const float4*)srcB)[1]);
  }
  __syncthreads();

  const int NIT = DD / 32;
  for (int it = 0; it < NIT; ++it) {
    const int  buf  = it & 1;
    const bool more = (it + 1) < NIT;

    v16h areg;
    float4 br0, br1;
    if (more) {
      areg = *(const v16h*)(srcA + (it + 1) * 32);
      const float4* pb = (const float4*)(srcB + (it + 1) * 32);
      br0 = pb[0]; br1 = pb[1];
    }

    v16h af[2], bf[2];
    #pragma unroll
    for (int mi = 0; mi < 2; ++mi) {
      const _Float16* p = Al[buf] + (wm * 32 + mi * 16 + ln) * 32 + g * 16;
      v8h lo = *(const v8h*)(p);
      v8h hi = *(const v8h*)(p + 8);
      af[mi] = __builtin_shufflevector(lo, hi, 0,1,2,3,4,5,6,7,8,9,10,11,12,13,14,15);
    }
    #pragma unroll
    for (int ni = 0; ni < 2; ++ni)
      bf[ni] = *(const v16h*)(Bl[buf] + (wn * 32 + ni * 16 + ln) * 32 + g * 16);
    #pragma unroll
    for (int mi = 0; mi < 2; ++mi)
      #pragma unroll
      for (int ni = 0; ni < 2; ++ni)
        acc[mi][ni] = wmma16(af[mi], bf[ni], acc[mi][ni]);

    if (more) {
      *(v16h*)(Al[buf ^ 1] + arow * 32 + acolb) = areg;
      *(v8h*) (Bl[buf ^ 1] + brow * 32 + bcolb) = cvt8(br0, br1);
    }
    __syncthreads();
  }

  #pragma unroll
  for (int ni = 0; ni < 2; ++ni) {
    const int d    = n0 + wn * 32 + ni * 16 + ln;
    const float bv = bo[d];
    #pragma unroll
    for (int mi = 0; mi < 2; ++mi) {
      #pragma unroll
      for (int v = 0; v < 8; ++v) {
        const int row = m0 + wm * 32 + mi * 16 + v + 8 * g;
        out[(size_t)row * DD + d] = acc[mi][ni][v] + bv;
      }
    }
  }
}

// ---------------------------------------------------------------------------
extern "C" void kernel_launch(void* const* d_in, const int* in_sizes, int n_in,
                              void* d_out, int out_size, void* d_ws, size_t ws_size,
                              hipStream_t stream) {
  (void)in_sizes; (void)n_in; (void)out_size; (void)ws_size;
  const float* x    = (const float*)d_in[0];
  const float* mask = (const float*)d_in[1];
  const float* Wq   = (const float*)d_in[2];
  const float* Wk   = (const float*)d_in[3];
  const float* Wv   = (const float*)d_in[4];
  const float* Wo   = (const float*)d_in[5];
  const float* bo   = (const float*)d_in[6];
  float* out = (float*)d_out;

  const size_t qkv_elems = (size_t)BH * LL * DH;     // 4,194,304 halfs each
  _Float16* Qh = (_Float16*)d_ws;
  _Float16* Kh = Qh + qkv_elems;
  _Float16* Vh = Kh + qkv_elems;
  _Float16* Hh = Vh + qkv_elems;                     // hidden [B*L, 1024] f16

  qkv_gemm<<<dim3(512, 1, 3), 256, 0, stream>>>(x, Wq, Wk, Wv, Qh, Kh, Vh);
  attn_kernel<<<dim3(LL / 128, BH), 256, 0, stream>>>(Qh, Kh, Vh, mask, Hh);
  out_gemm<<<dim3(512), 256, 0, stream>>>(Hh, Wo, bo, out);
}
